// ClinicalTrialEncoder_27212912787552
// MI455X (gfx1250) — matrix operand order repaired and checked by reading, not compiled, not run on hardware
//
#include <hip/hip_runtime.h>
#include <hip/hip_bf16.h>

typedef _Float16 h16;
typedef __attribute__((ext_vector_type(16))) _Float16 v16h;
typedef __attribute__((ext_vector_type(8)))  _Float16 v8h;
typedef __attribute__((ext_vector_type(8)))  float    v8f;

#define Bx 64
#define Sx 512
#define EDIM 256
#define Hx 256
#define G4 1024   /* 4*H */
#define NTAG 9

// ---------------------------------------------------------------------------
// WMMA fragment loaders (layouts per CDNA5 ISA §7.12.2)
// ---------------------------------------------------------------------------
// A-matrix 16x32 f16: lane L holds row m = L%16.
//   lanes 0-15 : elems[0..7]=K kb+0..7,  elems[8..15]=K kb+16..23
//   lanes 16-31: elems[0..7]=K kb+8..15, elems[8..15]=K kb+24..31
__device__ __forceinline__ v16h load_a_frag(const h16* row, int kb, int lane) {
  int base = kb + ((lane & 16) ? 8 : 0);
  v8h lo = *(const v8h*)(row + base);
  v8h hi = *(const v8h*)(row + base + 16);
  return __builtin_shufflevector(lo, hi, 0,1,2,3,4,5,6,7,8,9,10,11,12,13,14,15);
}

// B-matrix 32x16 f16 (KxN): lane L holds column n = L%16; same K order as A:
//   lanes 0-15 : elems[0..7]=K kb+0..7,  elems[8..15]=K kb+16..23
//   lanes 16-31: elems[0..7]=K kb+8..15, elems[8..15]=K kb+24..31
__device__ __forceinline__ v16h load_b_frag(const h16* wrow, int kb, int lane) {
  int base = kb + ((lane & 16) ? 8 : 0);
  v8h lo = *(const v8h*)(wrow + base);
  v8h hi = *(const v8h*)(wrow + base + 16);
  return __builtin_shufflevector(lo, hi, 0,1,2,3,4,5,6,7,8,9,10,11,12,13,14,15);
}

__device__ __forceinline__ float sigf(float x) { return 1.0f / (1.0f + __expf(-x)); }

// Async copy one 16B chunk global->LDS (gfx1250 VGLOBAL async path, ASYNCcnt).
__device__ __forceinline__ void async_b128(unsigned lds_off, const void* gptr) {
  unsigned long long ga = (unsigned long long)(size_t)gptr;
  asm volatile("global_load_async_to_lds_b128 %0, %1, off"
               :: "v"(lds_off), "v"(ga) : "memory");
}
__device__ __forceinline__ void wait_async0() {
  asm volatile("s_wait_asynccnt 0x0" ::: "memory");
}

// ---------------------------------------------------------------------------
// Small prep kernels
// ---------------------------------------------------------------------------
__global__ void cast_f32_to_f16(const float* __restrict__ src, h16* __restrict__ dst, int n) {
  int i = blockIdx.x * blockDim.x + threadIdx.x;
  if (i < n) dst[i] = (h16)src[i];
}

__global__ void bias_sum_kernel(const float* __restrict__ a, const float* __restrict__ b,
                                float* __restrict__ o, int n) {
  int i = blockIdx.x * blockDim.x + threadIdx.x;
  if (i < n) o[i] = a[i] + b[i];
}

__global__ void zero_out_kernel(float* __restrict__ o) { o[0] = 0.0f; }

// Embedding gather + f16 cast: xh[bs][e] = (f16) emb[ids[bs]][e]
__global__ void __launch_bounds__(256)
embed_kernel(const int* __restrict__ ids, const float* __restrict__ emb,
             h16* __restrict__ xh) {
  size_t idx = (size_t)blockIdx.x * 256 + threadIdx.x;   // B*S*EDIM total
  int bs = (int)(idx >> 8);
  int e  = (int)(idx & 255);
  int row = ids[bs];
  xh[idx] = (h16)emb[(size_t)row * EDIM + e];
}

// ---------------------------------------------------------------------------
// Input GEMM: xg[bs][g] = sum_e xh[bs][e] * w_ih[g][e] + (b_ih+b_hh)[g]
// M=32768, N=1024, K=256. Block = 8 waves; the shared 16x256 A tile is staged
// in LDS once per block with async copies (all 8 waves reuse it); each wave
// computes a 16(M)x64(N) strip. Output stored f16 (halves recurrence traffic).
// ---------------------------------------------------------------------------
__global__ void __launch_bounds__(256)
input_gemm_kernel(const h16* __restrict__ xh, const h16* __restrict__ w,
                  const float* __restrict__ bsum, h16* __restrict__ xg) {
  __shared__ h16 a_tile[16 * EDIM];            // 8KB at LDS offset 0
  int tid  = threadIdx.x;
  int lane = tid & 31;
  int wave = tid >> 5;                          // 0..7
  int m0 = blockIdx.x * 16;
  int n0 = blockIdx.y * 512 + wave * 64;
  int nlane = lane & 15;

  // stage A tile: 16 rows x 512B = 512 x b128 chunks, 2 per thread
  #pragma unroll
  for (int rep = 0; rep < 2; ++rep) {
    int chunk = tid + rep * 256;                // 0..511
    int r = chunk >> 5;                         // 32 x 16B chunks per row
    int cc = (chunk & 31) * 8;                  // half index within row
    unsigned lds_off = (unsigned)(r * EDIM + cc) * 2u;
    async_b128(lds_off, xh + (size_t)(m0 + r) * EDIM + cc);
  }
  wait_async0();
  __syncthreads();

  const h16* arow = a_tile + (size_t)nlane * EDIM;
  v8f acc[4] = {};
  #pragma unroll
  for (int kb = 0; kb < EDIM; kb += 32) {
    v16h a = load_a_frag(arow, kb, lane);       // from LDS
    #pragma unroll
    for (int sub = 0; sub < 4; ++sub) {
      const h16* brow = w + (size_t)(n0 + sub * 16 + nlane) * EDIM;
      v16h b = load_b_frag(brow, kb, lane);
      acc[sub] = __builtin_amdgcn_wmma_f32_16x16x32_f16(
          false, a, false, b, (short)0, acc[sub], false, false);
    }
  }
  int hi = (lane & 16) ? 8 : 0;
  #pragma unroll
  for (int sub = 0; sub < 4; ++sub) {
    int col = n0 + sub * 16 + nlane;
    float bs = bsum[col];
    #pragma unroll
    for (int v = 0; v < 8; ++v) {
      int row = m0 + v + hi;
      xg[(size_t)row * G4 + col] = (h16)(acc[sub][v] + bs);
    }
  }
}

// ---------------------------------------------------------------------------
// LSTM recurrence, one direction. Grid = 4 blocks (16 batch rows each),
// 512 threads = 16 waves, one block per WGP. h lives in LDS (f16), c in
// registers. Wave w owns hidden-unit columns [w*16,w*16+16) and computes its
// own i/f/g/o gate tiles (cols j, 256+j, 512+j, 768+j), so the cell update is
// wave-local. The next step's 32KB xg tile is async-copied into LDS while the
// current step's WMMAs run (single buffer; prefetch issued only after the
// barrier that ends consumption of the previous tile).
// ---------------------------------------------------------------------------
__global__ void __launch_bounds__(512)
lstm_rec_kernel(const h16* __restrict__ xg, const h16* __restrict__ whh,
                float* __restrict__ hout, int forward) {
  __shared__ h16 smem[16 * Hx + 16 * G4];       // h (8KB) + xg tile (32KB)
  h16* h_lds = smem;                            // [16][Hx], LDS offset 0
  h16* xtile = smem + 16 * Hx;                  // [16][G4], LDS offset 8192
  const unsigned XTILE_OFF = 16u * Hx * 2u;

  int tid  = threadIdx.x;
  int lane = tid & 31;
  int wave = tid >> 5;                          // 0..15
  int b0 = blockIdx.x * 16;
  int nlane = lane & 15;
  int hi = (lane & 16) ? 8 : 0;
  int j = wave * 16 + nlane;                    // hidden unit column (0..255)

  for (int idx = tid; idx < 16 * Hx; idx += 512) h_lds[idx] = (h16)0.0f;
  v8f c = {};

  // async prefetch of a step's gate tile: 16 rows x 2KB = 2048 chunks, 4/thread
  auto issue_tile = [&](int s) {
    #pragma unroll
    for (int rep = 0; rep < 4; ++rep) {
      int chunk = tid + rep * 512;              // 0..2047
      int r  = chunk >> 7;                      // 128 x 16B chunks per row
      int cc = (chunk & 127) * 8;               // half index within row
      unsigned lds_off = XTILE_OFF + (unsigned)(r * G4 + cc) * 2u;
      async_b128(lds_off, xg + ((size_t)(b0 + r) * Sx + s) * G4 + cc);
    }
  };
  issue_tile(forward ? 0 : (Sx - 1));
  __syncthreads();                              // h zeros visible

  for (int t = 0; t < Sx; ++t) {
    int s = forward ? t : (Sx - 1 - t);
    v8f acc[4] = {};
    const h16* arow = h_lds + (size_t)nlane * Hx;
    #pragma unroll
    for (int kb = 0; kb < Hx; kb += 32) {
      v16h a = load_a_frag(arow, kb, lane);     // from LDS
      #pragma unroll
      for (int g = 0; g < 4; ++g) {
        const h16* brow = whh + (size_t)(g * Hx + j) * Hx;   // w_hh[col][k]
        v16h bf = load_b_frag(brow, kb, lane);
        acc[g] = __builtin_amdgcn_wmma_f32_16x16x32_f16(
            false, a, false, bf, (short)0, acc[g], false, false);
      }
    }
    wait_async0();        // tile(t) landed in LDS
    __syncthreads();      // all waves done reading old h; tile visible

    #pragma unroll
    for (int v = 0; v < 8; ++v) {
      int m = v + hi;
      const h16* gt = xtile + (size_t)m * G4 + j;
      float gi = acc[0][v] + (float)gt[0 * Hx];
      float gf = acc[1][v] + (float)gt[1 * Hx];
      float gg = acc[2][v] + (float)gt[2 * Hx];
      float go = acc[3][v] + (float)gt[3 * Hx];
      float cn = sigf(gf) * c[v] + sigf(gi) * tanhf(gg);
      float hn = sigf(go) * tanhf(cn);
      c[v] = cn;
      h_lds[(size_t)m * Hx + j] = (h16)hn;
      hout[((size_t)(b0 + m) * Sx + s) * Hx + j] = hn;
    }
    __syncthreads();      // tile consumed, new h visible
    if (t + 1 < Sx) issue_tile(forward ? (t + 1) : (Sx - 2 - t));
  }
}

// ---------------------------------------------------------------------------
// Projection: feats[bs][tag] = [hf|hb] . w_out[tag] + b_out[tag]  (N=9, VALU)
// ---------------------------------------------------------------------------
__global__ void __launch_bounds__(256)
proj_kernel(const float* __restrict__ hf, const float* __restrict__ hb,
            const float* __restrict__ wout, const float* __restrict__ bout,
            float* __restrict__ feats) {
  int idx = blockIdx.x * 256 + threadIdx.x;
  if (idx >= Bx * Sx * NTAG) return;
  int tag = idx % NTAG;
  int bs  = idx / NTAG;
  const float* hr  = hf + (size_t)bs * Hx;
  const float* hbr = hb + (size_t)bs * Hx;
  const float* w1  = wout + (size_t)tag * (2 * Hx);
  const float* w2  = w1 + Hx;
  float acc = bout[tag];
  for (int k = 0; k < Hx; k += 4) {
    float4 a  = *(const float4*)(hr + k);
    float4 wa = *(const float4*)(w1 + k);
    acc += a.x * wa.x + a.y * wa.y + a.z * wa.z + a.w * wa.w;
    float4 b  = *(const float4*)(hbr + k);
    float4 wb = *(const float4*)(w2 + k);
    acc += b.x * wb.x + b.y * wb.y + b.z * wb.z + b.w * wb.w;
  }
  feats[idx] = acc;
}

// ---------------------------------------------------------------------------
// CRF NLL: one wave per batch row, lanes 0..8 hold alpha; logsumexp via shfl.
// ---------------------------------------------------------------------------
__global__ void __launch_bounds__(32)
crf_kernel(const float* __restrict__ feats, const int* __restrict__ tags,
           const unsigned char* __restrict__ mask,
           const float* __restrict__ start_t, const float* __restrict__ end_t,
           const float* __restrict__ trans, float* __restrict__ out) {
  int b = blockIdx.x;
  int lane = threadIdx.x;
  const float* em = feats + (size_t)b * Sx * NTAG;
  const int* tg = tags + b * Sx;
  const unsigned char* mk = mask + b * Sx;
  bool act = lane < NTAG;

  float tcol[NTAG];
  #pragma unroll
  for (int i = 0; i < NTAG; ++i)
    tcol[i] = act ? trans[i * NTAG + lane] : 0.0f;

  float alpha = act ? (start_t[lane] + em[lane]) : -1e30f;

  int tprev = tg[0];
  float score = start_t[tprev] + em[tprev];
  int seq_end = 0;

  for (int t = 1; t < Sx; ++t) {
    bool mt = mk[t] != 0;
    float mval = -1e30f;
    float vals[NTAG];
    #pragma unroll
    for (int i = 0; i < NTAG; ++i) {
      float ai = __shfl(alpha, i, 32);
      float v = ai + tcol[i];
      vals[i] = v;
      mval = fmaxf(mval, v);
    }
    float ssum = 0.0f;
    #pragma unroll
    for (int i = 0; i < NTAG; ++i) ssum += __expf(vals[i] - mval);
    float emt = act ? em[(size_t)t * NTAG + lane] : 0.0f;
    float na = mval + __logf(ssum) + emt;
    if (mt) { alpha = act ? na : alpha; seq_end = t; }

    int tt = tg[t];
    score += (trans[tprev * NTAG + tt] + em[(size_t)t * NTAG + tt]) * (mt ? 1.0f : 0.0f);
    tprev = tt;
  }
  int last_tag = tg[seq_end];
  score += end_t[last_tag];

  float av = act ? (alpha + end_t[lane]) : -1e30f;
  float mv = av;
  #pragma unroll
  for (int i = 0; i < NTAG; ++i) mv = fmaxf(mv, __shfl(av, i, 32));
  float ssum = 0.0f;
  #pragma unroll
  for (int i = 0; i < NTAG; ++i) ssum += __expf(__shfl(av, i, 32) - mv);
  float logZ = mv + __logf(ssum);

  if (lane == 0) {
    float llh = score - logZ;
    atomicAdd(out, -llh / (float)Bx);
  }
}

// ---------------------------------------------------------------------------
extern "C" void kernel_launch(void* const* d_in, const int* in_sizes, int n_in,
                              void* d_out, int out_size, void* d_ws, size_t ws_size,
                              hipStream_t stream) {
  const int*   sent    = (const int*)d_in[0];
  const int*   tags    = (const int*)d_in[1];
  const unsigned char* mask = (const unsigned char*)d_in[2];
  const float* emb     = (const float*)d_in[3];
  const float* w_ih_f  = (const float*)d_in[4];
  const float* w_hh_f  = (const float*)d_in[5];
  const float* b_ih_f  = (const float*)d_in[6];
  const float* b_hh_f  = (const float*)d_in[7];
  const float* w_ih_b  = (const float*)d_in[8];
  const float* w_hh_b  = (const float*)d_in[9];
  const float* b_ih_b  = (const float*)d_in[10];
  const float* b_hh_b  = (const float*)d_in[11];
  const float* w_out   = (const float*)d_in[12];
  const float* b_out   = (const float*)d_in[13];
  const float* start_t = (const float*)d_in[14];
  const float* end_t   = (const float*)d_in[15];
  const float* trans   = (const float*)d_in[16];
  float* outp = (float*)d_out;

  char* ws = (char*)d_ws;
  size_t off = 0;
  auto alloc = [&](size_t bytes) -> char* {
    char* p = ws + off;
    off = (off + bytes + 255) & ~(size_t)255;
    return p;
  };
  const size_t NW = (size_t)G4 * EDIM;           // 1024*256 weight elems
  h16*   xh     = (h16*)  alloc((size_t)Bx * Sx * EDIM * 2);
  h16*   wihf16 = (h16*)  alloc(NW * 2);
  h16*   wihb16 = (h16*)  alloc(NW * 2);
  h16*   whhf16 = (h16*)  alloc(NW * 2);
  h16*   whhb16 = (h16*)  alloc(NW * 2);
  float* bsum_f = (float*)alloc(G4 * 4);
  float* bsum_b = (float*)alloc(G4 * 4);
  h16*   xg_f   = (h16*)  alloc((size_t)Bx * Sx * G4 * 2);
  h16*   xg_b   = (h16*)  alloc((size_t)Bx * Sx * G4 * 2);
  float* hf     = (float*)alloc((size_t)Bx * Sx * Hx * 4);
  float* hb     = (float*)alloc((size_t)Bx * Sx * Hx * 4);
  float* feats  = (float*)alloc((size_t)Bx * Sx * NTAG * 4);

  // 1) weight casts + fused biases
  cast_f32_to_f16<<<(int)(NW / 256), 256, 0, stream>>>(w_ih_f, wihf16, (int)NW);
  cast_f32_to_f16<<<(int)(NW / 256), 256, 0, stream>>>(w_ih_b, wihb16, (int)NW);
  cast_f32_to_f16<<<(int)(NW / 256), 256, 0, stream>>>(w_hh_f, whhf16, (int)NW);
  cast_f32_to_f16<<<(int)(NW / 256), 256, 0, stream>>>(w_hh_b, whhb16, (int)NW);
  bias_sum_kernel<<<4, 256, 0, stream>>>(b_ih_f, b_hh_f, bsum_f, G4);
  bias_sum_kernel<<<4, 256, 0, stream>>>(b_ih_b, b_hh_b, bsum_b, G4);

  // 2) embedding gather
  embed_kernel<<<(Bx * Sx * EDIM) / 256, 256, 0, stream>>>(sent, emb, xh);

  // 3) input GEMMs (WMMA + async LDS staging), both directions
  dim3 ggrid(Bx * Sx / 16, G4 / 512);
  input_gemm_kernel<<<ggrid, 256, 0, stream>>>(xh, wihf16, bsum_f, xg_f);
  input_gemm_kernel<<<ggrid, 256, 0, stream>>>(xh, wihb16, bsum_b, xg_b);

  // 4) recurrences (WMMA from LDS-resident h, async-prefetched gate tiles)
  lstm_rec_kernel<<<Bx / 16, 512, 0, stream>>>(xg_f, whhf16, hf, 1);
  lstm_rec_kernel<<<Bx / 16, 512, 0, stream>>>(xg_b, whhb16, hb, 0);

  // 5) projection to 9 tags
  proj_kernel<<<(Bx * Sx * NTAG + 255) / 256, 256, 0, stream>>>(hf, hb, w_out, b_out, feats);

  // 6) CRF NLL -> scalar
  zero_out_kernel<<<1, 1, 0, stream>>>(outp);
  crf_kernel<<<Bx, 32, 0, stream>>>(feats, tags, mask, start_t, end_t, trans, outp);
}
